// GraphNetWithAttributes_70626442215730
// MI455X (gfx1250) — matrix operand order repaired and checked by reading, not compiled, not run on hardware
//
#include <hip/hip_runtime.h>
#include <hip/hip_bf16.h>
#include <math.h>

// ---------------- problem constants ----------------
#define N_NODES   50000
#define N_EDGES   800000
#define N_GRAPHS  50
#define D_IN      64
#define NB        10
#define FC        100
#define MAX_RADIUS 4.0f

#define EGROUPS   (N_EDGES / 16)     // 50000 (exact)
#define NGROUPS   (N_NODES / 16)     // 3125  (exact)

#define INV_SQRT_NB  0.31622776601683794f   // 1/sqrt(10)
#define INV_SQRT_FC  0.1f                   // 1/sqrt(100)
#define INV_SQRT_NN  0.25f                  // 1/sqrt(16)
#define INV_SQRT_NODES 0.03162277660168379f // 1/sqrt(1000)

// ---------------- WMMA types ----------------
typedef __attribute__((ext_vector_type(16))) _Float16 v16h;
typedef __attribute__((ext_vector_type(8)))  float    v8f;
typedef __attribute__((ext_vector_type(4)))  int      v4i;

union Frag16 {
    v16h v;
    _Float16 h[16];
    unsigned u[8];
};

__device__ __forceinline__ v8f wmma_f16(v16h a, v16h b, v8f c) {
    // D = A(16x32 f16) * B(32x16 f16) + C(16x16 f32)
    return __builtin_amdgcn_wmma_f32_16x16x32_f16(
        /*neg_a=*/false, a, /*neg_b=*/false, b,
        /*c_mod=*/(short)0, c, /*reuse_a=*/false, /*reuse_b=*/false);
}

__device__ __forceinline__ float silu_f(float x) {
    return x / (1.0f + __expf(-x));
}

// A-fragment (16x32 f16) element->K mapping for element e in lane L:
//   K = e + (L>=16 ? 16 : 0)
// For element pair (2v, 2v+1):  k0 = 2v + (L>=16 ? 16 : 0)
__device__ __forceinline__ int a_k0(int v, int hs) {
    return 2 * v + (hs ? 16 : 0);
}

// B-fragment (32x16 f16) element->K mapping for element e in lane L:
//   K = e + (L>=16 ? 16 : 0),  N = L & 15

// packed-weight layout per layer (offsets in halves)
#define WOFF_WA   0
#define WOFF_WB   3584                 // 7 frags * 512
#define WOFF_W1   (3584 + 16384)
#define WOFF_WSC  (3584 + 2*16384)
#define WOFF_W2   (3584 + 3*16384)
#define WSTRIDE_H (3584 + 4*16384)     // 69120 halves per layer
#define EDGE_WLDS_H (3584 + 16384)     // Wa + Wb staged region, 19968 halves = 39936 B

// ---------------- kernels ----------------

__global__ void zero_f32_kernel(float* __restrict__ p, int n) {
    int i = blockIdx.x * blockDim.x + threadIdx.x;
    int stride = gridDim.x * blockDim.x;
    for (; i < n; i += stride) p[i] = 0.0f;
}

__global__ void init_h_kernel(const float* __restrict__ x, _Float16* __restrict__ h0) {
    int i = blockIdx.x * blockDim.x + threadIdx.x;        // N*64 threads
    int n = i >> 6, c = i & 63;
    if (n < N_NODES) h0[(size_t)n * 128 + c] = (_Float16)x[(size_t)n * 64 + c];
}

// Pack all weights of one layer into f16 B-fragment layout (zero-padded).
__global__ void repack_kernel(const float* __restrict__ Wsc, const float* __restrict__ W1,
                              const float* __restrict__ Wa,  const float* __restrict__ Wb,
                              const float* __restrict__ W2,  _Float16* __restrict__ wbase,
                              int di, int dout) {
    int tid = blockIdx.x * blockDim.x + threadIdx.x;
    int stride = gridDim.x * blockDim.x;
    int NTd = di >> 4, NTo = dout >> 4, ktd = di >> 5;

    // Wa: [10,100] -> padded [32,112], 7 N-tiles, scale by 1/sqrt(NB)
    for (int idx = tid; idx < 7 * 512; idx += stride) {
        int nt = idx >> 9, r = idx & 511, lane = r >> 4, e = r & 15;
        int K = e + ((lane >= 16) ? 16 : 0);
        int Nn = nt * 16 + (lane & 15);
        float v = (K < NB && Nn < FC) ? Wa[K * FC + Nn] * INV_SQRT_NB : 0.0f;
        wbase[WOFF_WA + idx] = (_Float16)v;
    }
    // Wb: [100,di] -> padded [128,di], scale by 1/sqrt(FC)
    for (int idx = tid; idx < 4 * NTd * 512; idx += stride) {
        int kt = idx / (NTd * 512); int rem = idx - kt * NTd * 512;
        int nt = rem >> 9, r = rem & 511, lane = r >> 4, e = r & 15;
        int K = kt * 32 + e + ((lane >= 16) ? 16 : 0);
        int Nn = nt * 16 + (lane & 15);
        float v = (K < FC) ? Wb[K * di + Nn] * INV_SQRT_FC : 0.0f;
        wbase[WOFF_WB + idx] = (_Float16)v;
    }
    // W1: [di,di]
    for (int idx = tid; idx < ktd * NTd * 512; idx += stride) {
        int kt = idx / (NTd * 512); int rem = idx - kt * NTd * 512;
        int nt = rem >> 9, r = rem & 511, lane = r >> 4, e = r & 15;
        int K = kt * 32 + e + ((lane >= 16) ? 16 : 0);
        int Nn = nt * 16 + (lane & 15);
        wbase[WOFF_W1 + idx] = (_Float16)W1[K * di + Nn];
    }
    // Wsc: [di,dout]
    for (int idx = tid; idx < ktd * NTo * 512; idx += stride) {
        int kt = idx / (NTo * 512); int rem = idx - kt * NTo * 512;
        int nt = rem >> 9, r = rem & 511, lane = r >> 4, e = r & 15;
        int K = kt * 32 + e + ((lane >= 16) ? 16 : 0);
        int Nn = nt * 16 + (lane & 15);
        wbase[WOFF_WSC + idx] = (_Float16)Wsc[K * dout + Nn];
    }
    // W2: [di,dout], fold 1/sqrt(num_neighbors)
    for (int idx = tid; idx < ktd * NTo * 512; idx += stride) {
        int kt = idx / (NTo * 512); int rem = idx - kt * NTo * 512;
        int nt = rem >> 9, r = rem & 511, lane = r >> 4, e = r & 15;
        int K = kt * 32 + e + ((lane >= 16) ? 16 : 0);
        int Nn = nt * 16 + (lane & 15);
        wbase[WOFF_W2 + idx] = (_Float16)(W2[K * dout + Nn] * INV_SQRT_NN);
    }
}

// Radial embedding, written directly in A-fragment layout (K padded 10->32).
__global__ void emb_kernel(const float* __restrict__ pos, const int* __restrict__ esrc,
                           const int* __restrict__ edst, _Float16* __restrict__ embA) {
    int wave = threadIdx.x >> 5, lane = threadIdx.x & 31;
    int g = blockIdx.x * 4 + wave;                 // 12500 blocks * 4 = EGROUPS exact
    int m = lane & 15, hs = lane >> 4;
    int ed = g * 16 + m;
    int s = esrc[ed], d = edst[ed];
    float dx = pos[s * 3 + 0] - pos[d * 3 + 0];
    float dy = pos[s * 3 + 1] - pos[d * 3 + 1];
    float dz = pos[s * 3 + 2] - pos[d * 3 + 2];
    float r = sqrtf(dx * dx + dy * dy + dz * dz);

    const float step = MAX_RADIUS / (float)(NB + 1);
    const float C = 1.14136f * __expf(2.0f) * 3.1622776601683795f; // * sqrt(NB)
    Frag16 A;
#pragma unroll
    for (int e = 0; e < 16; ++e) {
        int K = e + (hs ? 16 : 0);
        float val = 0.0f;
        if (K < NB) {
            float center = (float)(K + 1) * step;
            float diff = (r - center) / step;
            float t1 = diff + 1.0f, t2 = 1.0f - diff;
            float s1 = (t1 > 0.0f) ? __expf(-1.0f / fmaxf(t1, 1e-9f)) : 0.0f;
            float s2 = (t2 > 0.0f) ? __expf(-1.0f / fmaxf(t2, 1e-9f)) : 0.0f;
            val = C * s1 * s2;
        }
        A.h[e] = (_Float16)val;
    }
    *(v16h*)(embA + (size_t)g * 512 + lane * 16) = A.v;
}

// y = h @ W1 (f16 out);  sc = (h * node_attr) @ Wsc (f32 out)
__global__ void node_pre_kernel(const _Float16* __restrict__ h, const float* __restrict__ node_attr,
                                const _Float16* __restrict__ wbase,
                                _Float16* __restrict__ y16, float* __restrict__ sc32,
                                int di, int dout) {
    int wave = threadIdx.x >> 5, lane = threadIdx.x & 31;
    int g = blockIdx.x * 4 + wave;
    if (g >= NGROUPS) return;                       // wave-uniform
    int m = lane & 15, hs = lane >> 4;
    int row = g * 16 + m;
    int ktiles = di >> 5, NTd = di >> 4, NTo = dout >> 4;
    const _Float16* W1P  = wbase + WOFF_W1;
    const _Float16* WscP = wbase + WOFF_WSC;

    float na = node_attr[row];
    Frag16 A[4], Ana[4];
    for (int kt = 0; kt < ktiles; ++kt) {
        const _Float16* base = h + (size_t)row * 128 + kt * 32;
#pragma unroll
        for (int v = 0; v < 8; ++v)
            A[kt].u[v] = *(const unsigned*)(base + a_k0(v, hs));
#pragma unroll
        for (int e = 0; e < 16; ++e)
            Ana[kt].h[e] = (_Float16)((float)A[kt].h[e] * na);
    }
    for (int nt = 0; nt < NTd; ++nt) {
        v8f acc = {};
        for (int kt = 0; kt < ktiles; ++kt) {
            Frag16 B; B.v = *(const v16h*)(W1P + ((size_t)(kt * NTd + nt)) * 512 + lane * 16);
            acc = wmma_f16(A[kt].v, B.v, acc);
        }
#pragma unroll
        for (int v = 0; v < 8; ++v) {
            int r = g * 16 + v + (hs ? 8 : 0);
            y16[(size_t)r * 128 + nt * 16 + m] = (_Float16)acc[v];
        }
    }
    for (int nt = 0; nt < NTo; ++nt) {
        v8f acc = {};
        for (int kt = 0; kt < ktiles; ++kt) {
            Frag16 B; B.v = *(const v16h*)(WscP + ((size_t)(kt * NTo + nt)) * 512 + lane * 16);
            acc = wmma_f16(Ana[kt].v, B.v, acc);
        }
#pragma unroll
        for (int v = 0; v < 8; ++v) {
            int r = g * 16 + v + (hs ? 8 : 0);
            sc32[(size_t)r * 128 + nt * 16 + m] = acc[v];
        }
    }
}

// Fused edge pipeline: stage Wa/Wb to LDS (async) -> GEMM1 -> silu -> LDS -> GEMM2
//                      -> gather y[edge_src] -> atomic scatter into agg[edge_dst]
__global__ void edge_kernel(const _Float16* __restrict__ embA, const _Float16* __restrict__ wbase,
                            const _Float16* __restrict__ y16, const float* __restrict__ edge_attr,
                            const int* __restrict__ esrc, const int* __restrict__ edst,
                            float* __restrict__ agg, int di) {
    __shared__ _Float16 wlds[EDGE_WLDS_H];          // Wa(7 frags) + Wb(<=32 frags) = 39.9 KB
    __shared__ _Float16 tile[4][16 * 128];          // per-wave 16x128 f16 tile (4 KB each)
    int wave = threadIdx.x >> 5, lane = threadIdx.x & 31;
    int tid = threadIdx.x;
    int g = blockIdx.x * 4 + wave;                  // 12500 blocks exact
    int m = lane & 15, hs = lane >> 4;
    _Float16* T = tile[wave];
    int NT2 = di >> 4;
    int total_halves = 3584 + 4 * NT2 * 512;        // Wa + Wb actually used

    // ---- stage packed Wa+Wb into LDS, shared by all 4 waves ----
#if __has_builtin(__builtin_amdgcn_global_load_async_to_lds_b128)
    {
        int nchunks = total_halves >> 3;            // 16-byte chunks
        for (int c = tid; c < nchunks; c += 128) {
            __builtin_amdgcn_global_load_async_to_lds_b128(
                (__attribute__((address_space(1))) v4i*)(wbase + (size_t)c * 8),
                (__attribute__((address_space(3))) v4i*)(wlds + (size_t)c * 8),
                /*offset=*/0, /*cpol=*/0);
        }
        asm volatile("s_wait_asynccnt 0x0" ::: "memory");
    }
#else
    for (int c = tid * 8; c < total_halves; c += 128 * 8)
        *(float4*)(wlds + c) = *(const float4*)(wbase + c);
#endif

    // per-lane edge metadata for the epilogue (row M = v + hs*8 of this group)
    int s8[8], d8[8]; float ea8[8];
#pragma unroll
    for (int v = 0; v < 8; ++v) {
        int ed = g * 16 + v + hs * 8;
        s8[v] = esrc[ed]; d8[v] = edst[ed]; ea8[v] = edge_attr[ed];
    }

    // zero pad region cols [112,128) of the silu tile
#pragma unroll
    for (int v = 0; v < 8; ++v) {
        int r = v + hs * 8;
        T[r * 128 + 112 + m] = (_Float16)0.0f;
    }
    __syncthreads();                                // weights staged + pad visible

    // GEMM1: emb[16x32] @ Wa[32x112], silu, stage f16 into LDS tile
    Frag16 A1; A1.v = *(const v16h*)(embA + (size_t)g * 512 + lane * 16);
    for (int nt = 0; nt < 7; ++nt) {
        Frag16 B; B.v = *(const v16h*)(wlds + WOFF_WA + nt * 512 + lane * 16);
        v8f acc = {};
        acc = wmma_f16(A1.v, B.v, acc);
#pragma unroll
        for (int v = 0; v < 8; ++v) {
            int r = v + hs * 8;
            T[r * 128 + nt * 16 + m] = (_Float16)silu_f(acc[v]);
        }
    }
    __syncthreads();

    // GEMM2: tile[16x128] @ Wb[128xdi] + epilogue (gather y, * edge_attr, scatter)
    for (int nt = 0; nt < NT2; ++nt) {
        v8f acc = {};
#pragma unroll
        for (int kt = 0; kt < 4; ++kt) {
            Frag16 A2;
#pragma unroll
            for (int v = 0; v < 8; ++v)
                A2.u[v] = *(const unsigned*)(&T[m * 128 + kt * 32 + a_k0(v, hs)]);
            Frag16 B; B.v = *(const v16h*)(wlds + WOFF_WB + ((size_t)(kt * NT2 + nt)) * 512 + lane * 16);
            acc = wmma_f16(A2.v, B.v, acc);
        }
        int col = nt * 16 + m;
#pragma unroll
        for (int v = 0; v < 8; ++v) {
            float yv = (float)y16[(size_t)s8[v] * 128 + col];
            atomicAdd(agg + (size_t)d8[v] * 128 + col, acc[v] * ea8[v] * yv);
        }
    }
}

// h_next = sc + (agg * node_attr) @ W2P   (1/sqrt(NN) folded into W2P), optional silu
__global__ void node_post_kernel(const float* __restrict__ agg, const float* __restrict__ sc32,
                                 const float* __restrict__ node_attr,
                                 const _Float16* __restrict__ wbase, _Float16* __restrict__ hnext,
                                 int di, int dout, int do_silu) {
    int wave = threadIdx.x >> 5, lane = threadIdx.x & 31;
    int g = blockIdx.x * 4 + wave;
    if (g >= NGROUPS) return;
    int m = lane & 15, hs = lane >> 4;
    int row = g * 16 + m;
    int ktiles = di >> 5, NTo = dout >> 4;
    const _Float16* W2P = wbase + WOFF_W2;

    float na = node_attr[row];
    Frag16 A[4];
    for (int kt = 0; kt < ktiles; ++kt) {
        const float* base = agg + (size_t)row * 128 + kt * 32;
#pragma unroll
        for (int v = 0; v < 8; ++v) {
            float2 f = *(const float2*)(base + a_k0(v, hs));
            A[kt].h[2 * v]     = (_Float16)(f.x * na);
            A[kt].h[2 * v + 1] = (_Float16)(f.y * na);
        }
    }
    for (int nt = 0; nt < NTo; ++nt) {
        v8f acc = {};
        for (int kt = 0; kt < ktiles; ++kt) {
            Frag16 B; B.v = *(const v16h*)(W2P + ((size_t)(kt * NTo + nt)) * 512 + lane * 16);
            acc = wmma_f16(A[kt].v, B.v, acc);
        }
#pragma unroll
        for (int v = 0; v < 8; ++v) {
            int r = g * 16 + v + (hs ? 8 : 0);
            float val = sc32[(size_t)r * 128 + nt * 16 + m] + acc[v];
            if (do_silu) val = silu_f(val);
            hnext[(size_t)r * 128 + nt * 16 + m] = (_Float16)val;
        }
    }
}

__global__ void pool_kernel(const _Float16* __restrict__ h, const int* __restrict__ batch,
                            float* __restrict__ out) {
    int i = blockIdx.x * blockDim.x + threadIdx.x;        // N*64 threads
    int n = i >> 6, c = i & 63;
    if (n < N_NODES) {
        int gph = batch[n];
        float v = (float)h[(size_t)n * 128 + c] * INV_SQRT_NODES;
        atomicAdd(out + gph * 64 + c, v);
    }
}

// ---------------- host launch ----------------
extern "C" void kernel_launch(void* const* d_in, const int* in_sizes, int n_in,
                              void* d_out, int out_size, void* d_ws, size_t ws_size,
                              hipStream_t stream) {
    (void)n_in; (void)out_size; (void)ws_size;

    // Input mapping: insertion order vs sorted (jax pytree) order, detected from sizes.
    const float *pos, *x, *node_attr, *edge_attr;
    const int *edge_src, *edge_dst, *batch;
    const float *Wsc[4], *W1[4], *Wa[4], *Wb[4], *W2[4];
    bool insertion = (in_sizes[0] == N_NODES * 3);
    if (insertion) {
        pos = (const float*)d_in[0]; x = (const float*)d_in[1];
        node_attr = (const float*)d_in[2]; edge_attr = (const float*)d_in[3];
        for (int l = 0; l < 4; ++l) {
            int b = 4 + 5 * l;
            Wsc[l] = (const float*)d_in[b + 0]; W1[l] = (const float*)d_in[b + 1];
            Wa[l]  = (const float*)d_in[b + 2]; Wb[l] = (const float*)d_in[b + 3];
            W2[l]  = (const float*)d_in[b + 4];
        }
        edge_src = (const int*)d_in[24]; edge_dst = (const int*)d_in[25];
        batch = (const int*)d_in[26];
    } else {
        batch = (const int*)d_in[0]; edge_attr = (const float*)d_in[1];
        edge_dst = (const int*)d_in[2]; edge_src = (const int*)d_in[3];
        node_attr = (const float*)d_in[4];
        for (int l = 0; l < 4; ++l) {
            int b = 5 + 5 * l;
            W1[l] = (const float*)d_in[b + 0]; W2[l] = (const float*)d_in[b + 1];
            Wa[l] = (const float*)d_in[b + 2]; Wb[l] = (const float*)d_in[b + 3];
            Wsc[l] = (const float*)d_in[b + 4];
        }
        pos = (const float*)d_in[25]; x = (const float*)d_in[26];
    }

    // workspace layout (all offsets 256B aligned); total ~141.4 MB
    char* ws = (char*)d_ws;
    _Float16* embA = (_Float16*)(ws + 0);            // EGROUPS*512 f16 = 51.2 MB
    _Float16* y16  = (_Float16*)(ws + 51200000);     // N*128 f16
    _Float16* h0   = (_Float16*)(ws + 64000000);     // N*128 f16
    _Float16* h1   = (_Float16*)(ws + 76800000);     // N*128 f16
    float*    sc32 = (float*)  (ws + 89600000);      // N*128 f32
    float*    agg  = (float*)  (ws + 115200000);     // N*128 f32
    _Float16* wpk  = (_Float16*)(ws + 140800000);    // 4 * WSTRIDE_H f16

    const int DI[4] = {64, 128, 128, 128};
    const int DO[4] = {128, 128, 128, 64};

    // 1) repack weights (per layer)
    for (int l = 0; l < 4; ++l)
        repack_kernel<<<128, 256, 0, stream>>>(Wsc[l], W1[l], Wa[l], Wb[l], W2[l],
                                               wpk + (size_t)l * WSTRIDE_H, DI[l], DO[l]);
    // 2) init h from x; zero out
    init_h_kernel<<<(N_NODES * 64 + 255) / 256, 256, 0, stream>>>(x, h0);
    zero_f32_kernel<<<64, 256, 0, stream>>>((float*)d_out, N_GRAPHS * 64);
    // 3) radial embedding in A-fragment layout
    emb_kernel<<<EGROUPS / 4, 128, 0, stream>>>(pos, edge_src, edge_dst, embA);

    _Float16* hcur = h0;
    _Float16* hnxt = h1;
    for (int l = 0; l < 4; ++l) {
        const _Float16* wb = wpk + (size_t)l * WSTRIDE_H;
        zero_f32_kernel<<<4096, 256, 0, stream>>>(agg, N_NODES * 128);
        node_pre_kernel<<<(NGROUPS + 3) / 4, 128, 0, stream>>>(hcur, node_attr, wb,
                                                               y16, sc32, DI[l], DO[l]);
        edge_kernel<<<EGROUPS / 4, 128, 0, stream>>>(embA, wb, y16, edge_attr,
                                                     edge_src, edge_dst, agg, DI[l]);
        node_post_kernel<<<(NGROUPS + 3) / 4, 128, 0, stream>>>(agg, sc32, node_attr, wb,
                                                                hnxt, DI[l], DO[l],
                                                                (l < 3) ? 1 : 0);
        _Float16* t = hcur; hcur = hnxt; hnxt = t;
    }
    // 4) pool per graph
    pool_kernel<<<(N_NODES * 64 + 255) / 256, 256, 0, stream>>>(hcur, batch, (float*)d_out);
}
